// FusedMoELinear_12086037971139
// MI455X (gfx1250) — matrix operand order repaired and checked by reading, not compiled, not run on hardware
//
#include <hip/hip_runtime.h>

// Problem constants (match reference)
#define E_    8
#define NTOK_ 4096
#define KD_   2048      // hidden dim K
#define ND_   2816      // intermediate dim N
#define P_    (NTOK_ * 2)
#define PE_   (P_ / E_) // 1024 pairs per expert

typedef __bf16 bf16_t;
typedef __attribute__((ext_vector_type(16))) __bf16 v16bf;
typedef __attribute__((ext_vector_type(8)))  __bf16 bf16x8;
typedef __attribute__((ext_vector_type(4)))  __bf16 bf16x4;
typedef __attribute__((ext_vector_type(8)))  float  v8f;
typedef uint32_t u32x4 __attribute__((ext_vector_type(4)));
typedef uint32_t u32x8 __attribute__((ext_vector_type(8)));
typedef __attribute__((address_space(3))) bf16_t* lds_bf16_ptr;

union FragA { v16bf v; bf16x8 h[2]; };

__device__ __forceinline__ bf16x4 cvt4(float4 f) {
  bf16x4 h;
  h[0] = (__bf16)f.x; h[1] = (__bf16)f.y; h[2] = (__bf16)f.z; h[3] = (__bf16)f.w;
  return h;
}

// TDM: async 2D tile load (bf16 elements) global -> LDS with row padding.
// D# packing per CDNA5 ISA 8.3/8.4. One descriptor pair (<=2D tensor).
// tile: tile_d0 elems per row, tile_d1 rows; row pitch d0_stride elems.
// LDS padding: pad 4 DWORDs (16B) after every 16 DWORDs (64B) -> 80B row pitch.
__device__ __forceinline__ void tdm_load_tile_bf16(uint32_t lds_byte_off,
                                                   const void* gaddr,
                                                   uint32_t tile_d0, uint32_t tile_d1,
                                                   uint32_t tensor_d0, uint32_t tensor_d1,
                                                   uint32_t d0_stride) {
  const uint64_t ga = (uint64_t)(uintptr_t)gaddr;
  u32x4 g0;
  g0[0] = 1u;                                   // count=1, user-mode, no gather
  g0[1] = lds_byte_off;                         // lds_addr
  g0[2] = (uint32_t)ga;                         // global_addr[31:0]
  g0[3] = ((uint32_t)(ga >> 32) & 0x01FFFFFFu)  // global_addr[56:32]
        | (2u << 30);                           // type=2 ("image")
  u32x8 g1;
  g1[0] = (1u << 16)                            // data_size=1 -> 2 bytes
        | (1u << 20)                            // pad_enable
        | (3u << 22)                            // pad_interval: 16 DWORDs
        | (3u << 25);                           // pad_amount: 4 DWORDs
  g1[1] = (tensor_d0 & 0xFFFFu) << 16;          // tensor_dim0[15:0] @ bits 63:48
  g1[2] = ((tensor_d0 >> 16) & 0xFFFFu)         // tensor_dim0[31:16]
        | ((tensor_d1 & 0xFFFFu) << 16);        // tensor_dim1[15:0]
  g1[3] = ((tensor_d1 >> 16) & 0xFFFFu)         // tensor_dim1[31:16]
        | (tile_d0 << 16);                      // tile_dim0
  g1[4] = tile_d1 & 0xFFFFu;                    // tile_dim1 (tile_dim2=0)
  g1[5] = d0_stride;                            // tensor_dim0_stride[31:0]
  g1[6] = 0u;                                   // stride hi, dim1_stride lo
  g1[7] = 0u;
  asm volatile("tensor_load_to_lds %0, %1" :: "s"(g0), "s"(g1) : "memory");
}

// ---------------------------------------------------------------------------
// Stage 1: H[p, n] = sigmoid(Xg . W1^T) * (Xg . W3^T), bf16 output to ws.
// Block tile: 256(M) x 64(N). 8 waves as 4(M) x 2(N); wave tile 64x32.
// ---------------------------------------------------------------------------
__global__ __launch_bounds__(256)
void moe_gateup_bf16(const float* __restrict__ x,
                     const float* __restrict__ W1,
                     const float* __restrict__ W3,
                     const int*   __restrict__ sids,
                     bf16_t*      __restrict__ H)
{
  constexpr int BM = 256, BN = 64, BK = 32, STR = 40; // STR pads LDS rows (80B)
  __shared__ bf16_t As [2][BM * STR];
  __shared__ bf16_t B1s[2][BN * STR];
  __shared__ bf16_t B3s[2][BN * STR];
  __shared__ int    tok[BM];

  const int e    = blockIdx.y;
  const int mb   = blockIdx.x & 3;    // PE_/BM == 4; fastest -> L2 weight reuse
  const int nb   = blockIdx.x >> 2;
  const int tid  = (int)threadIdx.x;
  const int lane = tid & 31;
  const int w    = tid >> 5;
  const int wm   = w >> 1;            // 0..3 -> M offset wm*64
  const int wn   = w & 1;             // 0..1 -> N offset wn*32
  const int lh   = lane >> 4;
  const int m16  = lane & 15;

  const int pbase = e * PE_ + mb * BM;
  const int nbase = nb * BN;
  const float* __restrict__ W1e = W1 + (size_t)e * ND_ * KD_ + (size_t)nbase * KD_;
  const float* __restrict__ W3e = W3 + (size_t)e * ND_ * KD_ + (size_t)nbase * KD_;

  tok[tid] = sids[pbase + tid];
  __syncthreads();

  const v8f zacc = {0.f,0.f,0.f,0.f,0.f,0.f,0.f,0.f};
  v8f accG[4][2] = {{zacc,zacc},{zacc,zacc},{zacc,zacc},{zacc,zacc}};
  v8f accU[4][2] = {{zacc,zacc},{zacc,zacc},{zacc,zacc},{zacc,zacc}};

  auto loadTile = [&](int kt, int buf) {
    const int k0 = kt * BK;
    #pragma unroll
    for (int i = 0; i < 8; ++i) {                 // A: 2048 float4 slots
      const int idx = tid + i * 256;
      const int r = idx >> 3, c = (idx & 7) * 4;
      const float4 f = *(const float4*)(x + (size_t)tok[r] * KD_ + (k0 + c));
      *(bf16x4*)&As[buf][r * STR + c] = cvt4(f);
    }
    #pragma unroll
    for (int i = 0; i < 2; ++i) {                 // B1/B3: 512 slots each
      const int idx = tid + i * 256;
      const int r = idx >> 3, c = (idx & 7) * 4;
      const float4 f1 = *(const float4*)(W1e + (size_t)r * KD_ + (k0 + c));
      *(bf16x4*)&B1s[buf][r * STR + c] = cvt4(f1);
      const float4 f3 = *(const float4*)(W3e + (size_t)r * KD_ + (k0 + c));
      *(bf16x4*)&B3s[buf][r * STR + c] = cvt4(f3);
      __builtin_prefetch(W1e + (size_t)r * KD_ + (k0 + c) + 2 * BK, 0, 0);
      __builtin_prefetch(W3e + (size_t)r * KD_ + (k0 + c) + 2 * BK, 0, 0);
    }
  };

  auto computeK = [&](int buf) {
    FragA a[4], b1[2], b3[2];
    #pragma unroll
    for (int mt = 0; mt < 4; ++mt) {
      const int row = wm * 64 + mt * 16 + m16;
      a[mt].h[0] = *(const bf16x8*)&As[buf][row * STR + lh * 16];
      a[mt].h[1] = *(const bf16x8*)&As[buf][row * STR + lh * 16 + 8];
    }
    #pragma unroll
    for (int nt = 0; nt < 2; ++nt) {
      const int col = wn * 32 + nt * 16 + m16;
      b1[nt].h[0] = *(const bf16x8*)&B1s[buf][col * STR + lh * 16];
      b1[nt].h[1] = *(const bf16x8*)&B1s[buf][col * STR + lh * 16 + 8];
      b3[nt].h[0] = *(const bf16x8*)&B3s[buf][col * STR + lh * 16];
      b3[nt].h[1] = *(const bf16x8*)&B3s[buf][col * STR + lh * 16 + 8];
    }
    #pragma unroll
    for (int mt = 0; mt < 4; ++mt)
      #pragma unroll
      for (int nt = 0; nt < 2; ++nt) {
        accG[mt][nt] = __builtin_amdgcn_wmma_f32_16x16x32_bf16(
            false, a[mt].v, false, b1[nt].v, (short)0, accG[mt][nt], false, false);
        accU[mt][nt] = __builtin_amdgcn_wmma_f32_16x16x32_bf16(
            false, a[mt].v, false, b3[nt].v, (short)0, accU[mt][nt], false, false);
      }
  };

  loadTile(0, 0);
  __syncthreads();
  constexpr int KT = KD_ / BK; // 64
  for (int kt = 0; kt < KT; ++kt) {
    const int cur = kt & 1;
    if (kt + 1 < KT) loadTile(kt + 1, cur ^ 1);
    computeK(cur);
    __syncthreads();
  }

  // Epilogue: h = up * sigmoid(gate); fast v_rcp_f32 path.
  #pragma unroll
  for (int mt = 0; mt < 4; ++mt)
    #pragma unroll
    for (int nt = 0; nt < 2; ++nt) {
      const int ncol = nbase + wn * 32 + nt * 16 + m16;
      #pragma unroll
      for (int v = 0; v < 8; ++v) {
        const float g  = accG[mt][nt][v];
        const float u  = accU[mt][nt][v];
        const float hv = u * __builtin_amdgcn_rcpf(1.0f + __expf(-g));
        const int prow = pbase + wm * 64 + mt * 16 + lh * 8 + v;
        H[(size_t)prow * ND_ + ncol] = (bf16_t)hv;
      }
    }
}

// ---------------------------------------------------------------------------
// Stage 2: out[token, k] += tw[p] * sum_n H[p,n] * W2[e][k,n]
// A tile (bf16 H rows) is moved by the Tensor Data Mover (async, TENSORcnt);
// B tile (fp32 W2) converted through VALU as before.
// ---------------------------------------------------------------------------
__global__ __launch_bounds__(256)
void moe_down_bf16(const bf16_t* __restrict__ H,
                   const float*  __restrict__ W2,
                   const float*  __restrict__ tw,
                   const int*    __restrict__ sids,
                   float*        __restrict__ out)
{
  constexpr int BM = 256, BN = 64, BK = 32, STR = 40;
  __shared__ bf16_t As[2][BM * STR];
  __shared__ bf16_t Bs[2][BN * STR];
  __shared__ int    tok[BM];
  __shared__ float  wgt[BM];

  const int e    = blockIdx.y;
  const int mb   = blockIdx.x & 3;
  const int kb   = blockIdx.x >> 2;
  const int tid  = (int)threadIdx.x;
  const int lane = tid & 31;
  const int w    = tid >> 5;
  const int wm   = w >> 1;
  const int wn   = w & 1;
  const int lh   = lane >> 4;
  const int m16  = lane & 15;

  const int pbase  = e * PE_ + mb * BM;
  const int kobase = kb * BN;
  const float*  __restrict__ W2e = W2 + (size_t)e * KD_ * ND_ + (size_t)kobase * ND_;
  const bf16_t* __restrict__ He  = H + (size_t)pbase * ND_;

  tok[tid] = sids[pbase + tid];
  wgt[tid] = tw[pbase + tid];
  __syncthreads();

  const v8f zacc = {0.f,0.f,0.f,0.f,0.f,0.f,0.f,0.f};
  v8f acc[4][2] = {{zacc,zacc},{zacc,zacc},{zacc,zacc},{zacc,zacc}};

  // A tile via TDM: 32 elems x 256 rows, row pitch ND_, into padded LDS rows.
  auto tdmLoadA = [&](int ns, int buf) {
    const uint32_t lds = (uint32_t)(uintptr_t)(lds_bf16_ptr)&As[buf][0];
    tdm_load_tile_bf16(lds, He + ns * BK,
                       /*tile_d0=*/BK, /*tile_d1=*/BM,
                       /*tensor_d0=*/ND_, /*tensor_d1=*/BM,
                       /*d0_stride=*/ND_);
  };

  auto loadB = [&](int ns, int buf) {
    const int n0 = ns * BK;
    #pragma unroll
    for (int i = 0; i < 2; ++i) {                 // B: 512 float4 slots
      const int idx = tid + i * 256;
      const int r = idx >> 3, c = (idx & 7) * 4;
      const float4 f = *(const float4*)(W2e + (size_t)r * ND_ + n0 + c);
      *(bf16x4*)&Bs[buf][r * STR + c] = cvt4(f);
      __builtin_prefetch(W2e + (size_t)r * ND_ + (n0 + c) + 2 * BK, 0, 0);
    }
  };

  auto computeK = [&](int buf) {
    FragA a[4], b[2];
    #pragma unroll
    for (int mt = 0; mt < 4; ++mt) {
      const int row = wm * 64 + mt * 16 + m16;
      a[mt].h[0] = *(const bf16x8*)&As[buf][row * STR + lh * 16];
      a[mt].h[1] = *(const bf16x8*)&As[buf][row * STR + lh * 16 + 8];
    }
    #pragma unroll
    for (int nt = 0; nt < 2; ++nt) {
      const int col = wn * 32 + nt * 16 + m16;
      b[nt].h[0] = *(const bf16x8*)&Bs[buf][col * STR + lh * 16];
      b[nt].h[1] = *(const bf16x8*)&Bs[buf][col * STR + lh * 16 + 8];
    }
    #pragma unroll
    for (int mt = 0; mt < 4; ++mt)
      #pragma unroll
      for (int nt = 0; nt < 2; ++nt)
        acc[mt][nt] = __builtin_amdgcn_wmma_f32_16x16x32_bf16(
            false, a[mt].v, false, b[nt].v, (short)0, acc[mt][nt], false, false);
  };

  if (w == 0) tdmLoadA(0, 0);                 // TDM ignores EXEC; wave-uniform issue
  loadB(0, 0);
  if (w == 0) __builtin_amdgcn_s_wait_tensorcnt(0);
  __syncthreads();

  constexpr int NT = ND_ / BK; // 88
  for (int s = 0; s < NT; ++s) {
    const int cur = s & 1;
    if (s + 1 < NT) {
      if (w == 0) tdmLoadA(s + 1, cur ^ 1);   // async while we compute `cur`
      loadB(s + 1, cur ^ 1);
    }
    computeK(cur);
    if (w == 0) __builtin_amdgcn_s_wait_tensorcnt(0); // drain before publishing
    __syncthreads();
  }

  // Epilogue: scale by router weight and atomically scatter-add per token
  #pragma unroll
  for (int mt = 0; mt < 4; ++mt)
    #pragma unroll
    for (int nt = 0; nt < 2; ++nt) {
      const int kcol = kobase + wn * 32 + nt * 16 + m16;
      #pragma unroll
      for (int v = 0; v < 8; ++v) {
        const int mrow = wm * 64 + mt * 16 + lh * 8 + v;
        const float val = acc[mt][nt][v] * wgt[mrow];
        atomicAdd(out + (size_t)tok[mrow] * KD_ + kcol, val);
      }
    }
}

// ---------------------------------------------------------------------------
__global__ void zero_f32(float4* __restrict__ p, int n4) {
  const int i = blockIdx.x * 256 + (int)threadIdx.x;
  if (i < n4) p[i] = make_float4(0.f, 0.f, 0.f, 0.f);
}

extern "C" void kernel_launch(void* const* d_in, const int* in_sizes, int n_in,
                              void* d_out, int out_size, void* d_ws, size_t ws_size,
                              hipStream_t stream) {
  (void)in_sizes; (void)n_in; (void)out_size; (void)ws_size;
  const float* x   = (const float*)d_in[0];
  const float* W1  = (const float*)d_in[1];
  const float* W3  = (const float*)d_in[2];
  const float* W2  = (const float*)d_in[3];
  const float* tw  = (const float*)d_in[4];
  const int*   sid = (const int*)d_in[5];
  // d_in[6] expert_ids: groups are [e*PE, (e+1)*PE) with expert id e -> implicit
  float*  out = (float*)d_out;
  bf16_t* H   = (bf16_t*)d_ws;   // 8192 x 2816 bf16 = 46.1 MB

  const int n4 = (NTOK_ * KD_) / 4;
  zero_f32<<<dim3((n4 + 255) / 256), dim3(256), 0, stream>>>((float4*)out, n4);

  moe_gateup_bf16<<<dim3(4 * (ND_ / 64), E_), dim3(256), 0, stream>>>(x, W1, W3, sid, H);
  moe_down_bf16 <<<dim3(4 * (KD_ / 64), E_), dim3(256), 0, stream>>>(H, W2, tw, sid, out);
}
